// BidirectionalRNNEncoder_32538672234955
// MI455X (gfx1250) — matrix operand order repaired and checked by reading, not compiled, not run on hardware
//
#include <hip/hip_runtime.h>
#include <hip/hip_bf16.h>
#include <stdint.h>

// ---------------------------------------------------------------------------
// BidirectionalRNNEncoder for MI455X (gfx1250)
//   phase 1: xproj = x @ Wx + b  via v_wmma_f32_16x16x32_bf16 (bf16x3 split)
//   phase 2: bidirectional tanh scan, Wh cached in LDS (TDM tensor_load_to_lds),
//            16 WGs per direction synced by a monotonic atomic barrier.
// ---------------------------------------------------------------------------

#define SEQ_LEN 16384
#define IDIM    1024
#define HDIM    1024

typedef __attribute__((ext_vector_type(16))) __bf16 v16bf;
typedef __attribute__((ext_vector_type(8)))  float  v8f;
typedef __attribute__((ext_vector_type(4)))  unsigned int u32x4;
typedef __attribute__((ext_vector_type(8)))  int    i32x8;
typedef __attribute__((ext_vector_type(4)))  int    i32x4;

// Split f32 into bf16 hi/lo (truncation split: f == hi + rest, lo ~= rest).
__device__ __forceinline__ void split_bf16(float f, __bf16& hi, __bf16& lo) {
    unsigned u = __builtin_bit_cast(unsigned, f);
    hi = __builtin_bit_cast(__bf16, (unsigned short)(u >> 16));
    float hf = __builtin_bit_cast(float, u & 0xFFFF0000u);
    float r  = f - hf;
    unsigned ur = __builtin_bit_cast(unsigned, r);
    lo = __builtin_bit_cast(__bf16, (unsigned short)(ur >> 16));
}

// ---------------------------------------------------------------------------
// Phase 1: xproj[s,j] = sum_k x[s,k]*Wx[k,j] + b[j]
// One 16x16 output tile per wave; 8 waves (8 j-tiles, same s-tile) per block.
// ---------------------------------------------------------------------------
#define GEMM_WAVES 8

__global__ void __launch_bounds__(256)
xproj_gemm_kernel(const float* __restrict__ x,
                  const float* __restrict__ Wx,
                  const float* __restrict__ bias,
                  float* __restrict__ xproj)
{
    const int lane  = threadIdx.x & 31;
    const int wave  = threadIdx.x >> 5;
    const int tile  = blockIdx.x * GEMM_WAVES + wave;   // 0 .. 65535
    const int sTile = tile >> 6;                        // 1024 s-tiles
    const int jTile = tile & 63;                        // 64 j-tiles
    const int s0    = sTile * 16;
    const int j0    = jTile * 16;

    const int mRow    = lane & 15;       // A row (M) for this lane
    const int hiHalf  = lane >> 4;       // 0: lanes 0-15, 1: lanes 16-31
    const int aKBase  = hiHalf * 8;      // A: K base per ISA 16-bit A layout
    const int bKBase  = hiHalf * 16;     // contiguous K base (unused; B is loaded with aKBase)
    const int nCol    = lane & 15;       // B/C/D column (N)

    v8f acc = {};
    const float* xrow = x + (size_t)(s0 + mRow) * IDIM;

    for (int k0 = 0; k0 < IDIM; k0 += 32) {
        v16bf ah, al, bh, bl;
#pragma unroll
        for (int e = 0; e < 16; ++e) {
            // A 16x32 bf16 layout: elem e -> K = aKBase + e (+8 if e>=8)
            int ka = aKBase + e + ((e >= 8) ? 8 : 0);
            __bf16 h, l;
            split_bf16(xrow[k0 + ka], h, l);
            ah[e] = h; al[e] = l;
            // B 32x16 bf16 layout: elem e -> K = aKBase + e (+8 if e>=8), col = nCol
            int kb = aKBase + e + ((e >= 8) ? 8 : 0);
            split_bf16(Wx[(size_t)(k0 + kb) * HDIM + j0 + nCol], h, l);
            bh[e] = h; bl[e] = l;
        }
        // bf16x3: hi*hi + hi*lo + lo*hi, all into f32 accumulator
        acc = __builtin_amdgcn_wmma_f32_16x16x32_bf16(false, ah, false, bh,
                                                      (short)0, acc, false, false);
        acc = __builtin_amdgcn_wmma_f32_16x16x32_bf16(false, ah, false, bl,
                                                      (short)0, acc, false, false);
        acc = __builtin_amdgcn_wmma_f32_16x16x32_bf16(false, al, false, bh,
                                                      (short)0, acc, false, false);
    }

    const float bv = bias[j0 + nCol];
#pragma unroll
    for (int v = 0; v < 8; ++v) {
        // C/D f32 16x16 layout: VGPR v -> M = v + 8*hiHalf, N = nCol
        int m = v + hiHalf * 8;
        xproj[(size_t)(s0 + m) * HDIM + j0 + nCol] = acc[v] + bv;
    }
}

// ---------------------------------------------------------------------------
// Init: zero h double-buffers and barrier counters (every call, deterministic)
// ---------------------------------------------------------------------------
__global__ void scan_init_kernel(float* __restrict__ hbuf, unsigned* __restrict__ cnt)
{
    int t = blockIdx.x * blockDim.x + threadIdx.x;
    if (t < 2 * 2 * HDIM) hbuf[t] = 0.0f;
    if (t < 2) cnt[t] = 0u;
}

// ---------------------------------------------------------------------------
// Phase 2: bidirectional scan. 32 blocks x 256 threads.
//   blockIdx.x / 16 = direction (0 fwd, 1 bwd); % 16 = column chunk (64 cols).
//   LDS: Wh slice [1024][64] f32 (256KB, loaded once via TDM) + h (4KB) + partials.
// ---------------------------------------------------------------------------
#define SCAN_THREADS 256
#define NCHUNKS      16
#define CHUNK_COLS   64

#if __has_builtin(__builtin_amdgcn_tensor_load_to_lds)
#define USE_TDM 1
#else
#define USE_TDM 0
#endif

__global__ void __launch_bounds__(SCAN_THREADS, 1)
rnn_scan_kernel(const float* __restrict__ xproj,
                const float* __restrict__ Wh,
                float* __restrict__ out,        // [2][SEQ_LEN][HDIM]
                float* __restrict__ hbuf,       // [2 dirs][2 bufs][HDIM]
                unsigned* __restrict__ cnt)     // [2 dirs]
{
    extern __shared__ float smem[];
    float* whs = smem;                               // [HDIM][CHUNK_COLS]
    float* hs  = smem + HDIM * CHUNK_COLS;           // [HDIM]
    float* ps  = hs + HDIM;                          // [SCAN_THREADS]

    const int tid  = threadIdx.x;
    const int dir  = blockIdx.x / NCHUNKS;           // 0 = forward, 1 = backward
    const int chk  = blockIdx.x % NCHUNKS;
    const int col0 = chk * CHUNK_COLS;

    // ---- Preload Wh[:, col0 : col0+64] into LDS ----
#if USE_TDM
    if (tid < 32) {  // one wave issues the TDM descriptor
        unsigned ldsOff = (unsigned)(uintptr_t)whs;  // low 32 bits of LDS-aperture addr
        unsigned long long ga = (unsigned long long)(uintptr_t)(Wh + col0);
        u32x4 g0;
        g0[0] = 1u;                                          // count = 1 (valid D#)
        g0[1] = ldsOff;                                      // lds_addr
        g0[2] = (unsigned)(ga & 0xFFFFFFFFull);              // global_addr[31:0]
        g0[3] = (unsigned)((ga >> 32) & 0x01FFFFFFull)       // global_addr[56:32]
              | (2u << 30);                                  // type = 2 ("image")
        i32x8 g1;
        g1[0] = (int)(2u << 16);                             // data_size = 4B
        g1[1] = (int)(((unsigned)HDIM & 0xFFFFu) << 16);     // tensor_dim0 lo16
        g1[2] = (int)(((unsigned)HDIM & 0xFFFFu) << 16);     // tensor_dim1 lo16
        g1[3] = (int)((unsigned)CHUNK_COLS << 16);           // tile_dim0 = 64
        g1[4] = (int)HDIM;                                   // tile_dim1 = 1024
        g1[5] = (int)HDIM;                                   // tensor_dim0_stride lo32
        g1[6] = 0;                                           // stride0 hi / stride1 lo16
        g1[7] = (int)((unsigned)(HDIM * HDIM) >> 16);        // stride1[47:16]
        i32x4 g2 = {0, 0, 0, 0};                             // groups 2/3 unused (2D)
        i32x4 g3 = {0, 0, 0, 0};
        i32x8 g4 = {0, 0, 0, 0, 0, 0, 0, 0};                 // 6-arg toolchain extra group
        __builtin_amdgcn_tensor_load_to_lds(g0, g1, g2, g3, g4, 0);
        __builtin_amdgcn_s_wait_tensorcnt(0);
    }
#else
    for (int idx = tid; idx < HDIM * CHUNK_COLS; idx += SCAN_THREADS)
        whs[idx] = Wh[(size_t)(idx >> 6) * HDIM + col0 + (idx & 63)];
#endif
    __syncthreads();

    float*    hread0  = hbuf + (size_t)dir * 2 * HDIM;
    unsigned* mycnt   = cnt + dir;
    float*    outbase = out + (size_t)dir * SEQ_LEN * HDIM;

    const int col = tid & (CHUNK_COLS - 1);
    const int seg = tid >> 6;                    // 0..3: rows [seg*256, seg*256+256)
    const float* wcol = whs + (size_t)seg * 256 * CHUNK_COLS + col;

    for (int step = 0; step < SEQ_LEN; ++step) {
        const float* hr = hread0 + (size_t)(step & 1) * HDIM;
        float*       hw = hread0 + (size_t)((step + 1) & 1) * HDIM;

        // Pull full h into LDS (agent-scope loads: bypass stale L0)
        for (int i = tid; i < HDIM; i += SCAN_THREADS)
            hs[i] = __hip_atomic_load(hr + i, __ATOMIC_RELAXED, __HIP_MEMORY_SCOPE_AGENT);
        __syncthreads();

        // Partial dot: 256 rows per segment, 64 columns
        float acc = 0.0f;
        const float* hseg = hs + seg * 256;
#pragma unroll 8
        for (int i = 0; i < 256; ++i)
            acc = fmaf(hseg[i], wcol[(size_t)i * CHUNK_COLS], acc);
        ps[tid] = acc;
        __syncthreads();

        if (tid < CHUNK_COLS) {
            float s = ps[tid] + ps[tid + 64] + ps[tid + 128] + ps[tid + 192];
            int srow = dir ? (SEQ_LEN - 1 - step) : step;
            float hn = tanhf(xproj[(size_t)srow * HDIM + col0 + tid] + s);
            outbase[(size_t)step * HDIM + col0 + tid] = hn;
            __hip_atomic_store(hw + col0 + tid, hn,
                               __ATOMIC_RELAXED, __HIP_MEMORY_SCOPE_AGENT);
        }
        __syncthreads();

        // Cross-WG barrier for this direction (monotonic counter; the release
        // half of the acq_rel RMW publishes our h stores, acquire half of the
        // spin load pulls the other WGs' stores).
        if (tid == 0) {
            __hip_atomic_fetch_add(mycnt, 1u, __ATOMIC_ACQ_REL, __HIP_MEMORY_SCOPE_AGENT);
            unsigned tgt = (unsigned)NCHUNKS * (unsigned)(step + 1);
            while (__hip_atomic_load(mycnt, __ATOMIC_ACQUIRE, __HIP_MEMORY_SCOPE_AGENT) < tgt)
                __builtin_amdgcn_s_sleep(1);
        }
        __syncthreads();
    }
}

// ---------------------------------------------------------------------------
// Launch
// ---------------------------------------------------------------------------
extern "C" void kernel_launch(void* const* d_in, const int* in_sizes, int n_in,
                              void* d_out, int out_size, void* d_ws, size_t ws_size,
                              hipStream_t stream)
{
    const float* x  = (const float*)d_in[0];  // [SEQ_LEN][IDIM]
    const float* Wx = (const float*)d_in[1];  // [IDIM][HDIM]
    const float* Wh = (const float*)d_in[2];  // [HDIM][HDIM]
    const float* b  = (const float*)d_in[3];  // [HDIM]
    float* out = (float*)d_out;               // [2][SEQ_LEN][HDIM]

    // Workspace layout (needs ~64.02 MB):
    //   [0, 64MB)  xproj f32 [SEQ_LEN][HDIM]
    //   then       h double buffers f32 [2][2][HDIM]
    //   then       barrier counters u32 [2]
    char* ws = (char*)d_ws;
    const size_t xprojBytes = (size_t)SEQ_LEN * HDIM * sizeof(float);
    float*    xproj = (float*)ws;
    float*    hbuf  = (float*)(ws + xprojBytes);
    unsigned* cnt   = (unsigned*)(ws + xprojBytes + (size_t)2 * 2 * HDIM * sizeof(float));

    // 1) reset scan state (deterministic across graph replays)
    scan_init_kernel<<<dim3(16), dim3(256), 0, stream>>>(hbuf, cnt);

    // 2) input projection GEMM (WMMA bf16x3)
    const int nTiles  = (SEQ_LEN / 16) * (HDIM / 16);        // 65536
    const int nBlocks = nTiles / GEMM_WAVES;                 // 8192
    xproj_gemm_kernel<<<dim3(nBlocks), dim3(256), 0, stream>>>(x, Wx, b, xproj);

    // 3) bidirectional scan, Wh resident in LDS (256KB slice per WG)
    const size_t shmem = (size_t)(HDIM * CHUNK_COLS + HDIM + SCAN_THREADS) * sizeof(float);
    (void)hipFuncSetAttribute((const void*)rnn_scan_kernel,
                              hipFuncAttributeMaxDynamicSharedMemorySize, (int)shmem);
    rnn_scan_kernel<<<dim3(2 * NCHUNKS), dim3(SCAN_THREADS), shmem, stream>>>(
        xproj, Wh, out, hbuf, cnt);
}
